// VLRModel_20186346291789
// MI455X (gfx1250) — matrix operand order repaired and checked by reading, not compiled, not run on hardware
//
#include <hip/hip_runtime.h>
#include <hip/hip_bf16.h>

typedef _Float16 v16h __attribute__((ext_vector_type(16)));
typedef _Float16 v8h  __attribute__((ext_vector_type(8)));
typedef float    v8f  __attribute__((ext_vector_type(8)));

#define BATCH   8
#define NPTS    4096
#define HID     256
#define INCH    194
#define KPAD    224      // 7 * 32
#define KSTEPS  7
#define TILE_N  16
#define FEAT_HW 65536    // 256*256
#define HEAD_HALFS (HID * KPAD)

// ---------------------------------------------------------------------------
// One-time weight repack: 4 heads of W1 (f32, 256x194) -> f16 (256x224, zero
// padded K) so WMMA A-fragments are two aligned 16B loads per k-step.
// ---------------------------------------------------------------------------
__global__ __launch_bounds__(256)
void vlr_pack_w1(const float* __restrict__ w0, const float* __restrict__ w1,
                 const float* __restrict__ w2, const float* __restrict__ w3,
                 _Float16* __restrict__ out) {
    int idx = blockIdx.x * 256 + threadIdx.x;
    if (idx >= 4 * HEAD_HALFS) return;
    int head = idx / HEAD_HALFS;
    int rem  = idx % HEAD_HALFS;
    int row  = rem / KPAD;
    int col  = rem % KPAD;
    const float* src = (head == 0) ? w0 : (head == 1) ? w1 : (head == 2) ? w2 : w3;
    float v = (col < INCH) ? src[row * INCH + col] : 0.0f;
    out[idx] = (_Float16)v;
}

// ---------------------------------------------------------------------------
// can_poly reduction: per-batch masked min over the 4096 points (x and y).
// ---------------------------------------------------------------------------
__global__ __launch_bounds__(256)
void vlr_minred(const float* __restrict__ py, const float* __restrict__ mask,
                float* __restrict__ mins) {
    __shared__ float sx[256];
    __shared__ float sy[256];
    const int b = blockIdx.x;
    const int t = threadIdx.x;
    float mnx = 1e9f, mny = 1e9f;
    for (int i = t; i < NPTS; i += 256) {
        int gp = b * NPTS + i;
        float mv = mask[gp];
        float x = py[gp * 2 + 0];
        float y = py[gp * 2 + 1];
        if (mv > 0.f) { mnx = fminf(mnx, x); mny = fminf(mny, y); }
    }
    sx[t] = mnx; sy[t] = mny;
    __syncthreads();
    for (int s = 128; s > 0; s >>= 1) {
        if (t < s) {
            sx[t] = fminf(sx[t], sx[t + s]);
            sy[t] = fminf(sy[t], sy[t + s]);
        }
        __syncthreads();
    }
    if (t == 0) { mins[b * 2 + 0] = sx[0]; mins[b * 2 + 1] = sy[0]; }
}

// ---------------------------------------------------------------------------
// One head: gather 16 points' 194-ch inputs into LDS (f16), WMMA GEMM1
// (256x224 x 224x16, f16 in / f32 accum), ReLU, tiny GEMM2, point update
// (mode 0) or cls + scaled-point outputs (mode 1).
// ---------------------------------------------------------------------------
__global__ __launch_bounds__(256)
void vlr_stage(const float* __restrict__ feat,
               const float* __restrict__ py_in,
               const float* __restrict__ mask,
               const _Float16* __restrict__ w1p,   // packed f16 [256][224]
               const float* __restrict__ b1,       // [256]
               const float* __restrict__ w2,       // [rows][256]
               const float* __restrict__ b2,       // [rows]
               const float* __restrict__ mins,     // [8][2] (mode 0 only)
               float* __restrict__ out0,           // mode0: py_next; mode1: py*4
               float* __restrict__ out1,           // mode1: cls
               int mode) {
    __shared__ __align__(32) _Float16 xs[KPAD][TILE_N];  // 7 KB
    __shared__ float hbuf[HID][TILE_N];                  // 16 KB
    __shared__ float w2s[2 * HID];                       // 2 KB
    __shared__ float xpt[TILE_N][2];
    __shared__ float msk[TILE_N];

    const int t    = threadIdx.x;
    const int b    = blockIdx.x >> 8;        // 256 point-tiles per batch
    const int tile = blockIdx.x & 255;
    const int p0   = tile * TILE_N;
    const int rows = (mode == 0) ? 2 : 1;

    if (t < TILE_N) {
        int gp = b * NPTS + p0 + t;
        xpt[t][0] = py_in[gp * 2 + 0];
        xpt[t][1] = py_in[gp * 2 + 1];
        msk[t]    = mask[gp];
    }
    for (int i = t; i < rows * HID; i += 256) w2s[i] = w2[i];
    __syncthreads();

    float minx = 0.f, miny = 0.f;
    if (mode == 0) { minx = mins[b * 2 + 0]; miny = mins[b * 2 + 1]; }

    // ---- build f16 input tile X[KPAD][16]: bilinear gather + coord channels
    for (int idx = t; idx < KPAD * TILE_N; idx += 256) {
        int p = idx & (TILE_N - 1);
        int c = idx >> 4;
        float pxr = xpt[p][0], pyr = xpt[p][1], mk = msk[p];
        float val = 0.f;
        if (c < 192) {
            // grid_sample(align_corners=True) net mapping: px * (W-1)/W
            float sx = pxr * (255.0f / 256.0f);
            float sy = pyr * (255.0f / 256.0f);
            float fx = floorf(sx), fy = floorf(sy);
            int x0 = (int)fx, y0 = (int)fy;
            float wx1 = sx - fx, wy1 = sy - fy;
            float wx0 = 1.f - wx1, wy0 = 1.f - wy1;
            const float* fc = feat + ((size_t)(b * 192 + c)) * FEAT_HW;
            float t00 = 0.f, t10 = 0.f, t01 = 0.f, t11 = 0.f;
            if ((unsigned)x0 < 256u && (unsigned)y0 < 256u)             t00 = fc[y0 * 256 + x0];
            if ((unsigned)(x0 + 1) < 256u && (unsigned)y0 < 256u)       t10 = fc[y0 * 256 + x0 + 1];
            if ((unsigned)x0 < 256u && (unsigned)(y0 + 1) < 256u)       t01 = fc[(y0 + 1) * 256 + x0];
            if ((unsigned)(x0 + 1) < 256u && (unsigned)(y0 + 1) < 256u) t11 = fc[(y0 + 1) * 256 + x0 + 1];
            val = (t00 * wx0 * wy0 + t10 * wx1 * wy0 +
                   t01 * wx0 * wy1 + t11 * wx1 * wy1) * mk;
        } else if (c == 192) {
            val = (mode == 0) ? (pxr - minx) * mk : pxr;  // canonical x / raw x
        } else if (c == 193) {
            val = (mode == 0) ? (pyr - miny) * mk : pyr;  // canonical y / raw y
        }
        xs[c][p] = (_Float16)val;
    }
    __syncthreads();

    // ---- GEMM1: 8 waves x 2 M-tiles = 256 hidden rows, K = 7 steps of 32
    const int wave  = t >> 5;          // wave32
    const int lane  = t & 31;
    const int mrow  = lane & 15;                 // A: M = lane&15
    const int khalf = (lane < 16) ? 0 : 8;       // A: K base within k-step
    const int mBase = (lane < 16) ? 0 : 8;       // C/D: M = vgpr + mBase
    const int m0a = wave * 16;
    const int m0b = (wave + 8) * 16;

    v8f acc0 = {};
    v8f acc1 = {};
#pragma unroll
    for (int k = 0; k < KSTEPS; ++k) {
        const int k0 = k * 32;
        // B fragment: lane holds column N = lane&15, K = k0+khalf+{0..7} and k0+16+khalf+{0..7}
        v16h bf;
#pragma unroll
        for (int i = 0; i < 8; ++i) {
            bf[i] = xs[k0 + khalf + i][mrow];
            bf[8 + i] = xs[k0 + 16 + khalf + i][mrow];
        }
        // A fragments: two contiguous 8-half runs per lane
        const _Float16* a0p = w1p + (size_t)(m0a + mrow) * KPAD + k0 + khalf;
        const _Float16* a1p = w1p + (size_t)(m0b + mrow) * KPAD + k0 + khalf;
        v8h a0lo = *(const v8h*)(a0p);
        v8h a0hi = *(const v8h*)(a0p + 16);
        v8h a1lo = *(const v8h*)(a1p);
        v8h a1hi = *(const v8h*)(a1p + 16);
        v16h a0, a1;
#pragma unroll
        for (int i = 0; i < 8; ++i) {
            a0[i] = a0lo[i]; a0[8 + i] = a0hi[i];
            a1[i] = a1lo[i]; a1[8 + i] = a1hi[i];
        }
        acc0 = __builtin_amdgcn_wmma_f32_16x16x32_f16(false, a0, false, bf,
                                                      (short)0, acc0, false, false);
        acc1 = __builtin_amdgcn_wmma_f32_16x16x32_f16(false, a1, false, bf,
                                                      (short)0, acc1, false, false);
    }

    // bias + ReLU -> LDS hidden buffer (f32)
#pragma unroll
    for (int v = 0; v < 8; ++v) {
        int h0 = m0a + mBase + v;
        int h1 = m0b + mBase + v;
        float x0v = acc0[v] + b1[h0];
        float x1v = acc1[v] + b1[h1];
        hbuf[h0][mrow] = x0v > 0.f ? x0v : 0.f;
        hbuf[h1][mrow] = x1v > 0.f ? x1v : 0.f;
    }
    __syncthreads();

    // ---- GEMM2 (2x256 or 1x256) + epilogue
    if (t < rows * TILE_N) {
        int p = t & 15;
        int r = t >> 4;
        float sum = b2[r];
#pragma unroll 4
        for (int h = 0; h < HID; ++h) sum += w2s[r * HID + h] * hbuf[h][p];
        if (mode == 0) {
            float off = sum * msk[p];
            int gp = b * NPTS + p0 + p;
            out0[gp * 2 + r] = xpt[p][r] + off * 4.0f;   // STRIDE
        } else {
            out1[b * NPTS + p0 + p] = sum * msk[p];       // cls
        }
    }
    if (mode == 1 && t >= 128 && t < 160) {
        int i = t - 128;
        int p = i >> 1, comp = i & 1;
        int gp = b * NPTS + p0 + p;
        out0[gp * 2 + comp] = xpt[p][comp] * 4.0f;        // py * DOWN_RATIO
    }
}

// ---------------------------------------------------------------------------
extern "C" void kernel_launch(void* const* d_in, const int* in_sizes, int n_in,
                              void* d_out, int out_size, void* d_ws, size_t ws_size,
                              hipStream_t stream) {
    const float* feat = (const float*)d_in[0];
    const float* pts  = (const float*)d_in[1];
    const float* mask = (const float*)d_in[2];
    // params order: w1_0,b1_0,w2_0,b2_0, w1_1,..., w1_2,..., w1c,b1c,w2c,b2c
    const float* w1[4] = {(const float*)d_in[3],  (const float*)d_in[7],
                          (const float*)d_in[11], (const float*)d_in[15]};
    const float* b1[4] = {(const float*)d_in[4],  (const float*)d_in[8],
                          (const float*)d_in[12], (const float*)d_in[16]};
    const float* w2[4] = {(const float*)d_in[5],  (const float*)d_in[9],
                          (const float*)d_in[13], (const float*)d_in[17]};
    const float* b2[4] = {(const float*)d_in[6],  (const float*)d_in[10],
                          (const float*)d_in[14], (const float*)d_in[18]};

    char* ws = (char*)d_ws;
    _Float16* w1pack = (_Float16*)ws;                 // 4*256*224*2 = 458752 B
    float* mins = (float*)(ws + 458752);              // 64 B
    float* pyA  = (float*)(ws + 458816);              // 262144 B
    float* pyB  = (float*)(ws + 720960);              // 262144 B

    vlr_pack_w1<<<(4 * HEAD_HALFS + 255) / 256, 256, 0, stream>>>(
        w1[0], w1[1], w1[2], w1[3], w1pack);

    const float* src = pts;
    float* dsts[3] = {pyA, pyB, pyA};
    for (int i = 0; i < 3; ++i) {
        vlr_minred<<<BATCH, 256, 0, stream>>>(src, mask, mins);
        vlr_stage<<<BATCH * 256, 256, 0, stream>>>(
            feat, src, mask,
            w1pack + (size_t)i * HEAD_HALFS, b1[i], w2[i], b2[i], mins,
            dsts[i], nullptr, /*mode=*/0);
        src = dsts[i];
    }

    float* out_py  = (float*)d_out;                    // (8,4096,2) * DOWN_RATIO
    float* out_cls = (float*)d_out + BATCH * NPTS * 2; // (8,4096)
    vlr_stage<<<BATCH * 256, 256, 0, stream>>>(
        feat, src, mask,
        w1pack + (size_t)3 * HEAD_HALFS, b1[3], w2[3], b2[3], mins,
        out_py, out_cls, /*mode=*/1);
}
